// Reason_58591943852604
// MI455X (gfx1250) — matrix operand order repaired and checked by reading, not compiled, not run on hardware
//
#include <hip/hip_runtime.h>
#include <hip/hip_bf16.h>
#include <math.h>

// ---------------- constants ----------------
#define BB      4096
#define RR      1000
#define DD      300
#define DP      320      // D padded to mult of 32
#define FIN     600
#define FP      608      // F_IN padded to mult of 32
#define LAT     512
#define N640    640      // Wd2 output padded 600 -> 640
#define TOPK    2
#define DISK    20
#define TEMPv   0.07f
#define CB      512                // batch chunk for neg path
#define NCH     (CB * DISK)        // 10240 neg rows per chunk
#define NCHUNK  (BB / CB)          // 8

typedef _Float16 v16h __attribute__((ext_vector_type(16)));
typedef float    v8f  __attribute__((ext_vector_type(8)));
typedef _Float16 h2   __attribute__((ext_vector_type(2)));

// ---------------- generic WMMA GEMM ----------------
// C[M x Np] = act(A[M x Kp](f16,row-major) @ Bt^T + bias), Bt is Np x Kp (W transposed)
// Block: 256 threads = 8 wave32s, block tile 64(M) x 128(N), each wave owns a 32x32 tile
// (4 accumulators; 2 A-frags + 2 B-frags -> 4 WMMAs per K-step).
// act: 0=none 1=tanh 2=sigmoid 3=relu. Ch (f16, ld=Np) and/or Cf (f32, ld=ldCf).
__global__ void __launch_bounds__(256)
gemm_f16_wmma(const _Float16* __restrict__ A,
              const _Float16* __restrict__ Bt,
              const float* __restrict__ bias, int Nbias,
              _Float16* __restrict__ Ch,
              float* __restrict__ Cf, int ldCf,
              int M, int Np, int Kp, int act)
{
    const int tid  = (int)threadIdx.x;
    const int wave = tid >> 5;
    const int lane = tid & 31;
    const int m16  = lane & 15;
    const int half = lane >> 4;

    const int waveM = (int)blockIdx.y * 64  + (wave >> 2) * 32;
    const int waveN = (int)blockIdx.x * 128 + (wave & 3) * 32;

    v8f acc00 = {}, acc01 = {}, acc10 = {}, acc11 = {};

    const _Float16* arow0 = A  + (size_t)(waveM + m16) * Kp;
    const _Float16* arow1 = arow0 + (size_t)16 * Kp;
    const _Float16* brow0 = Bt + (size_t)(waveN + m16) * Kp + half * 16;
    const _Float16* brow1 = brow0 + (size_t)16 * Kp;

    for (int k0 = 0; k0 < Kp; k0 += 32) {
        v16h af0 = {}, af1 = {}, bf0 = {}, bf1 = {};
        // B fragments: lane holds column n=lane%16 (+16), k = k0 + 2v + 16*half
        #pragma unroll
        for (int v = 0; v < 8; ++v) {
            h2 t0 = *(const h2*)(brow0 + k0 + 2 * v);
            h2 t1 = *(const h2*)(brow1 + k0 + 2 * v);
            bf0[2 * v] = t0.x; bf0[2 * v + 1] = t0.y;
            bf1[2 * v] = t1.x; bf1[2 * v + 1] = t1.y;
        }
        // A fragments: lane holds row m=lane%16 (+16), k = k0 + 2v + 16*half
        #pragma unroll
        for (int v = 0; v < 8; ++v) {
            const int kLo = 2 * v + 16 * half;
            h2 t0 = *(const h2*)(arow0 + k0 + kLo);
            h2 t1 = *(const h2*)(arow1 + k0 + kLo);
            af0[2 * v] = t0.x; af0[2 * v + 1] = t0.y;
            af1[2 * v] = t1.x; af1[2 * v + 1] = t1.y;
        }
        if (k0 + 32 < Kp) {   // lowers to global_prefetch_b8
            __builtin_prefetch(arow0 + k0 + 32, 0, 1);
            __builtin_prefetch(brow0 + k0 + 32, 0, 1);
        }
        acc00 = __builtin_amdgcn_wmma_f32_16x16x32_f16(false, af0, false, bf0, (short)0, acc00, false, false);
        acc01 = __builtin_amdgcn_wmma_f32_16x16x32_f16(false, af0, false, bf1, (short)0, acc01, false, false);
        acc10 = __builtin_amdgcn_wmma_f32_16x16x32_f16(false, af1, false, bf0, (short)0, acc10, false, false);
        acc11 = __builtin_amdgcn_wmma_f32_16x16x32_f16(false, af1, false, bf1, (short)0, acc11, false, false);
    }

    const int n0 = waveN + m16;
    const int n1 = n0 + 16;
    const float bv0 = (bias && n0 < Nbias) ? bias[n0] : 0.0f;
    const float bv1 = (bias && n1 < Nbias) ? bias[n1] : 0.0f;

    #pragma unroll
    for (int r = 0; r < 8; ++r) {
        int row0 = waveM + r + half * 8;   // C/D layout: vgpr r, lanes16-31 -> M=r+8
        int row1 = row0 + 16;
        float v00 = acc00[r] + bv0, v01 = acc01[r] + bv1;
        float v10 = acc10[r] + bv0, v11 = acc11[r] + bv1;
        if (act == 1) {
            v00 = tanhf(v00); v01 = tanhf(v01); v10 = tanhf(v10); v11 = tanhf(v11);
        } else if (act == 2) {
            v00 = 1.0f/(1.0f+expf(-v00)); v01 = 1.0f/(1.0f+expf(-v01));
            v10 = 1.0f/(1.0f+expf(-v10)); v11 = 1.0f/(1.0f+expf(-v11));
        } else if (act == 3) {
            v00 = fmaxf(v00,0.0f); v01 = fmaxf(v01,0.0f);
            v10 = fmaxf(v10,0.0f); v11 = fmaxf(v11,0.0f);
        }
        if (Ch) {
            Ch[(size_t)row0 * Np + n0] = (_Float16)v00; Ch[(size_t)row0 * Np + n1] = (_Float16)v01;
            Ch[(size_t)row1 * Np + n0] = (_Float16)v10; Ch[(size_t)row1 * Np + n1] = (_Float16)v11;
        }
        if (Cf) {
            Cf[(size_t)row0 * ldCf + n0] = v00; Cf[(size_t)row0 * ldCf + n1] = v01;
            Cf[(size_t)row1 * ldCf + n0] = v10; Cf[(size_t)row1 * ldCf + n1] = v11;
        }
    }
}

// ---------------- weight convert + transpose (+zero pad) ----------------
// W is K x N row-major f32 -> Wt is Np x Kp row-major f16 (Wt[n][k] = W[k][n])
__global__ void convert_wt_kernel(const float* __restrict__ W, _Float16* __restrict__ Wt,
                                  int K, int N, int Kp, int Np)
{
    int idx = blockIdx.x * 256 + threadIdx.x;
    if (idx >= Np * Kp) return;
    int n = idx / Kp, k = idx % Kp;
    float v = (n < N && k < K) ? W[(size_t)k * N + n] : 0.0f;
    Wt[idx] = (_Float16)v;
}

// ---------------- top-k / bottom-k with jax tie semantics ----------------
__device__ int arg_extreme(float* sv, float* rv, int* ri, int tid, int n,
                           bool findMax, int skip0, int skip1)
{
    __syncthreads();
    float bv = 0.0f; int bi = -1;
    for (int i = tid; i < n; i += 256) {
        if (i == skip0 || i == skip1) continue;
        float v = sv[i];
        if (bi < 0) { bv = v; bi = i; continue; }
        bool better = findMax ? (v > bv) : (v < bv);
        if (better) { bv = v; bi = i; }       // ties keep lower index (ascending scan)
    }
    rv[tid] = bv; ri[tid] = bi;
    __syncthreads();
    for (int s = 128; s > 0; s >>= 1) {
        if (tid < s) {
            float ov = rv[tid + s]; int oi = ri[tid + s];
            bool take;
            if (ri[tid] < 0)      take = true;
            else if (oi < 0)      take = false;
            else take = findMax ? (ov > rv[tid] || (ov == rv[tid] && oi < ri[tid]))
                                : (ov < rv[tid] || (ov == rv[tid] && oi < ri[tid]));
            if (take) { rv[tid] = ov; ri[tid] = oi; }
        }
        __syncthreads();
    }
    return ri[0];
}

__global__ void __launch_bounds__(256)
topk_kernel(const float* __restrict__ logits, const int* __restrict__ rels,
            int* __restrict__ pos_keys, int* __restrict__ neg_keys)
{
    __shared__ float sv[RR];
    __shared__ float rv[256];
    __shared__ int   ri[256];
    int b = blockIdx.x, tid = threadIdx.x;
    for (int i = tid; i < RR; i += 256) sv[i] = logits[(size_t)b * RR + i];
    __syncthreads();

    int i1 = arg_extreme(sv, rv, ri, tid, RR, true, -1, -1);
    int i2 = arg_extreme(sv, rv, ri, tid, RR, true, i1, -1);
    if (tid == 0) {
        pos_keys[b * 2 + 0] = rels[i1 * 2 + 1];
        pos_keys[b * 2 + 1] = rels[i2 * 2 + 1];
    }
    for (int t = 0; t < DISK; ++t) {
        int im = arg_extreme(sv, rv, ri, tid, RR, false, -1, -1);
        if (tid == 0) {
            neg_keys[b * DISK + t] = rels[im * 2 + 1];
            sv[im] = __builtin_inff();
        }
        __syncthreads();
    }
}

// ---------------- gathers (f32 -> f16, with K-padding) ----------------
__global__ void gather_x_kernel(const float* __restrict__ rel_emb,
                                const int* __restrict__ pos_keys, _Float16* __restrict__ Xh)
{
    int idx = blockIdx.x * 256 + threadIdx.x;
    if (idx >= BB * FP) return;
    int b = idx / FP, c = idx % FP;
    float v = 0.0f;
    if (c < FIN) {
        int key = pos_keys[b * 2 + (c >= DD ? 1 : 0)];
        v = rel_emb[(size_t)key * DD + (c % DD)];
    }
    Xh[idx] = (_Float16)v;
}

__global__ void gather_pos1_kernel(const float* __restrict__ rel_emb,
                                   const int* __restrict__ pos_keys, _Float16* __restrict__ P)
{
    int idx = blockIdx.x * 256 + threadIdx.x;
    if (idx >= BB * DP) return;
    int b = idx / DP, c = idx % DP;
    float v = (c < DD) ? rel_emb[(size_t)pos_keys[b * 2] * DD + c] : 0.0f;
    P[idx] = (_Float16)v;
}

__global__ void gather_neg_kernel(const float* __restrict__ rel_emb,
                                  const int* __restrict__ neg_keys,
                                  _Float16* __restrict__ negh, int r0)
{
    int idx = blockIdx.x * 256 + threadIdx.x;
    if (idx >= NCH * DP) return;
    int i = idx / DP, c = idx % DP;
    float v = (c < DD) ? rel_emb[(size_t)neg_keys[r0 + i] * DD + c] : 0.0f;
    negh[idx] = (_Float16)v;
}

// ---------------- neg logits: one wave32 per neg row ----------------
__global__ void __launch_bounds__(256)
neg_dot_kernel(const float* __restrict__ zp, const _Float16* __restrict__ negp,
               float* __restrict__ neg_logit, int r0)
{
    int wave = threadIdx.x >> 5, lane = threadIdx.x & 31;
    int i = blockIdx.x * 8 + wave;
    if (i >= NCH) return;
    int r = r0 + i;
    int b = r / DISK;
    const _Float16* np = negp + (size_t)i * LAT;
    const float*    z  = zp   + (size_t)b * LAT;
    float s = 0.0f;
    for (int d = lane; d < LAT; d += 32) s += z[d] * (float)np[d];
    #pragma unroll
    for (int o = 16; o > 0; o >>= 1) s += __shfl_down(s, o, 32);
    if (lane == 0) neg_logit[r] = s;
}

// ---------------- per-row losses ----------------
__global__ void __launch_bounds__(256)
vae_kernel(const float* __restrict__ rel_emb, const int* __restrict__ pos_keys,
           const float* __restrict__ xrec, float* __restrict__ part)
{
    __shared__ float red[256];
    int b = blockIdx.x, tid = threadIdx.x;
    float s = 0.0f;
    for (int f = tid; f < FIN; f += 256) {
        int key = pos_keys[b * 2 + (f >= DD ? 1 : 0)];
        float xv = rel_emb[(size_t)key * DD + (f % DD)];
        float d  = xv - xrec[(size_t)b * N640 + f];
        s += d * d;
    }
    red[tid] = s; __syncthreads();
    for (int st = 128; st > 0; st >>= 1) { if (tid < st) red[tid] += red[tid + st]; __syncthreads(); }
    if (tid == 0) part[b] = red[0] * (1.0f / (float)FIN);
}

__global__ void __launch_bounds__(256)
csl_kernel(const float* __restrict__ zp, const float* __restrict__ posvec,
           const float* __restrict__ neg_logit, float* __restrict__ part)
{
    __shared__ float red[256];
    int b = blockIdx.x, tid = threadIdx.x;
    float s = 0.0f;
    for (int d = tid; d < LAT; d += 256)
        s += zp[(size_t)b * LAT + d] * posvec[(size_t)b * LAT + d];
    red[tid] = s; __syncthreads();
    for (int st = 128; st > 0; st >>= 1) { if (tid < st) red[tid] += red[tid + st]; __syncthreads(); }
    if (tid == 0) {
        const float inv_t = 1.0f / TEMPv;
        float lg0 = red[0] * inv_t;
        float m = lg0;
        float lg[DISK];
        for (int n = 0; n < DISK; ++n) { lg[n] = neg_logit[b * DISK + n] * inv_t; m = fmaxf(m, lg[n]); }
        float sum = expf(lg0 - m);
        for (int n = 0; n < DISK; ++n) sum += expf(lg[n] - m);
        part[b] = (logf(sum) + m) - lg0;
    }
}

// deterministic final reduce (fixed order, no atomics)
__global__ void __launch_bounds__(256)
finalize_kernel(const float* __restrict__ vae_part, const float* __restrict__ csl_part,
                float* __restrict__ out)
{
    __shared__ float r0[256], r1[256];
    int tid = threadIdx.x;
    float a = 0.0f, c = 0.0f;
    for (int i = tid; i < BB; i += 256) { a += vae_part[i]; c += csl_part[i]; }
    r0[tid] = a; r1[tid] = c; __syncthreads();
    for (int st = 128; st > 0; st >>= 1) {
        if (tid < st) { r0[tid] += r0[tid + st]; r1[tid] += r1[tid + st]; }
        __syncthreads();
    }
    if (tid == 0) { out[0] = r0[0]; out[1] = r1[0]; }
}

__global__ void zero_row_kernel(float* __restrict__ out)
{
    int i = blockIdx.x * 256 + threadIdx.x;
    if (i < LAT) out[2 + i] = 0.0f;   // z_rel row 0 = zeros
}

// ---------------- host side ----------------
static inline int ceil_div(long long a, long long b) { return (int)((a + b - 1) / b); }

extern "C" void kernel_launch(void* const* d_in, const int* in_sizes, int n_in,
                              void* d_out, int out_size, void* d_ws, size_t ws_size,
                              hipStream_t stream)
{
    (void)in_sizes; (void)n_in; (void)out_size; (void)ws_size;

    const float* logits  = (const float*)d_in[0];
    const int*   rels    = (const int*)  d_in[1];
    const float* rel_emb = (const float*)d_in[2];
    const float* We1 = (const float*)d_in[3];  const float* be1 = (const float*)d_in[4];
    const float* We2 = (const float*)d_in[5];  const float* be2 = (const float*)d_in[6];
    const float* Wd1 = (const float*)d_in[7];  const float* bd1 = (const float*)d_in[8];
    const float* Wd2 = (const float*)d_in[9];  const float* bd2 = (const float*)d_in[10];
    const float* Wz1 = (const float*)d_in[11]; const float* bz1 = (const float*)d_in[12];
    const float* Wz2 = (const float*)d_in[13]; const float* bz2 = (const float*)d_in[14];
    const float* Wi1 = (const float*)d_in[15]; const float* bi1 = (const float*)d_in[16];
    const float* Wi2 = (const float*)d_in[17]; const float* bi2 = (const float*)d_in[18];
    // Wk* == Wi* by construction; reuse Wi conversions.

    float* out = (float*)d_out;
    float* zp  = out + 2 + LAT;        // z_rel rows 1..B  (B x 512, row-major)

    // workspace layout
    char* base = (char*)d_ws;
    size_t off = 0;
    auto alloc = [&](size_t bytes) -> char* {
        char* p = base + off;
        off = (off + bytes + 255) & ~(size_t)255;
        return p;
    };
    int*       pos_keys = (int*)alloc((size_t)BB * 2 * 4);
    int*       neg_keys = (int*)alloc((size_t)BB * DISK * 4);
    _Float16*  We1t = (_Float16*)alloc((size_t)LAT * FP * 2);
    _Float16*  We2t = (_Float16*)alloc((size_t)LAT * LAT * 2);
    _Float16*  Wd1t = (_Float16*)alloc((size_t)LAT * LAT * 2);
    _Float16*  Wd2t = (_Float16*)alloc((size_t)N640 * LAT * 2);
    _Float16*  Wz1t = (_Float16*)alloc((size_t)LAT * LAT * 2);
    _Float16*  Wz2t = (_Float16*)alloc((size_t)LAT * LAT * 2);
    _Float16*  Wi1t = (_Float16*)alloc((size_t)LAT * DP * 2);
    _Float16*  Wi2t = (_Float16*)alloc((size_t)LAT * LAT * 2);
    _Float16*  Xh    = (_Float16*)alloc((size_t)BB * FP * 2);
    _Float16*  pos1h = (_Float16*)alloc((size_t)BB * DP * 2);
    _Float16*  h1    = (_Float16*)alloc((size_t)BB * LAT * 2);
    _Float16*  zh    = (_Float16*)alloc((size_t)BB * LAT * 2);
    _Float16*  d1h   = (_Float16*)alloc((size_t)BB * LAT * 2);
    _Float16*  zp1h  = (_Float16*)alloc((size_t)BB * LAT * 2);
    _Float16*  posh  = (_Float16*)alloc((size_t)BB * LAT * 2);
    float*     xrec  = (float*)alloc((size_t)BB * N640 * 4);
    float*     posvec= (float*)alloc((size_t)BB * LAT * 4);
    _Float16*  negh  = (_Float16*)alloc((size_t)NCH * DP * 2);
    _Float16*  negHid= (_Float16*)alloc((size_t)NCH * LAT * 2);
    _Float16*  negp  = (_Float16*)alloc((size_t)NCH * LAT * 2);
    float*     neg_logit = (float*)alloc((size_t)BB * DISK * 4);
    float*     vae_part  = (float*)alloc((size_t)BB * 4);
    float*     csl_part  = (float*)alloc((size_t)BB * 4);

    // 0) z_rel zero row
    zero_row_kernel<<<2, 256, 0, stream>>>(out);

    // 1) weight convert+transpose (zero-padded K)
    convert_wt_kernel<<<ceil_div((long long)LAT * FP, 256), 256, 0, stream>>>(We1, We1t, FIN, LAT, FP, LAT);
    convert_wt_kernel<<<ceil_div((long long)LAT * LAT, 256), 256, 0, stream>>>(We2, We2t, LAT, LAT, LAT, LAT);
    convert_wt_kernel<<<ceil_div((long long)LAT * LAT, 256), 256, 0, stream>>>(Wd1, Wd1t, LAT, LAT, LAT, LAT);
    convert_wt_kernel<<<ceil_div((long long)N640 * LAT, 256), 256, 0, stream>>>(Wd2, Wd2t, LAT, FIN, LAT, N640);
    convert_wt_kernel<<<ceil_div((long long)LAT * LAT, 256), 256, 0, stream>>>(Wz1, Wz1t, LAT, LAT, LAT, LAT);
    convert_wt_kernel<<<ceil_div((long long)LAT * LAT, 256), 256, 0, stream>>>(Wz2, Wz2t, LAT, LAT, LAT, LAT);
    convert_wt_kernel<<<ceil_div((long long)LAT * DP, 256), 256, 0, stream>>>(Wi1, Wi1t, DD, LAT, DP, LAT);
    convert_wt_kernel<<<ceil_div((long long)LAT * LAT, 256), 256, 0, stream>>>(Wi2, Wi2t, LAT, LAT, LAT, LAT);

    // 2) top-2 / bottom-20 selection -> keys
    topk_kernel<<<BB, 256, 0, stream>>>(logits, rels, pos_keys, neg_keys);

    // 3) gathers
    gather_x_kernel<<<ceil_div((long long)BB * FP, 256), 256, 0, stream>>>(rel_emb, pos_keys, Xh);
    gather_pos1_kernel<<<ceil_div((long long)BB * DP, 256), 256, 0, stream>>>(rel_emb, pos_keys, pos1h);

    const dim3 blk(256);
    // 4) encoder: h1 = tanh(X @ We1 + be1); z = tanh(h1 @ We2 + be2)
    gemm_f16_wmma<<<dim3(LAT / 128, BB / 64), blk, 0, stream>>>(Xh, We1t, be1, LAT, h1, nullptr, 0, BB, LAT, FP, 1);
    gemm_f16_wmma<<<dim3(LAT / 128, BB / 64), blk, 0, stream>>>(h1, We2t, be2, LAT, zh, nullptr, 0, BB, LAT, LAT, 1);
    // 5) decoder: d1 = tanh(z @ Wd1 + bd1); x_rec = sigmoid(d1 @ Wd2 + bd2)
    gemm_f16_wmma<<<dim3(LAT / 128, BB / 64), blk, 0, stream>>>(zh, Wd1t, bd1, LAT, d1h, nullptr, 0, BB, LAT, LAT, 1);
    gemm_f16_wmma<<<dim3(N640 / 128, BB / 64), blk, 0, stream>>>(d1h, Wd2t, bd2, FIN, nullptr, xrec, N640, BB, N640, LAT, 2);
    // 6) VAE loss partials
    vae_kernel<<<BB, 256, 0, stream>>>(rel_emb, pos_keys, xrec, vae_part);
    // 7) projection head: z_p = relu(z @ Wz1 + bz1) @ Wz2 + bz2  -> written into d_out (z_rel rows 1..B)
    gemm_f16_wmma<<<dim3(LAT / 128, BB / 64), blk, 0, stream>>>(zh, Wz1t, bz1, LAT, zp1h, nullptr, 0, BB, LAT, LAT, 3);
    gemm_f16_wmma<<<dim3(LAT / 128, BB / 64), blk, 0, stream>>>(zp1h, Wz2t, bz2, LAT, nullptr, zp, LAT, BB, LAT, LAT, 0);
    // 8) positive path: pos_vec = relu(pos1 @ Wi1 + bi1) @ Wi2 + bi2
    gemm_f16_wmma<<<dim3(LAT / 128, BB / 64), blk, 0, stream>>>(pos1h, Wi1t, bi1, LAT, posh, nullptr, 0, BB, LAT, DP, 3);
    gemm_f16_wmma<<<dim3(LAT / 128, BB / 64), blk, 0, stream>>>(posh, Wi2t, bi2, LAT, nullptr, posvec, LAT, BB, LAT, LAT, 0);
    // 9) negative path in 8 chunks (Wk == Wi)
    for (int c = 0; c < NCHUNK; ++c) {
        int r0 = c * NCH;
        gather_neg_kernel<<<ceil_div((long long)NCH * DP, 256), 256, 0, stream>>>(rel_emb, neg_keys, negh, r0);
        gemm_f16_wmma<<<dim3(LAT / 128, NCH / 64), blk, 0, stream>>>(negh, Wi1t, bi1, LAT, negHid, nullptr, 0, NCH, LAT, DP, 3);
        gemm_f16_wmma<<<dim3(LAT / 128, NCH / 64), blk, 0, stream>>>(negHid, Wi2t, bi2, LAT, negp, nullptr, 0, NCH, LAT, LAT, 0);
        neg_dot_kernel<<<NCH / 8, 256, 0, stream>>>(zp, negp, neg_logit, r0);
    }
    // 10) contrastive loss partials + deterministic final reduce
    csl_kernel<<<BB, 256, 0, stream>>>(zp, posvec, neg_logit, csl_part);
    finalize_kernel<<<1, 256, 0, stream>>>(vae_part, csl_part, out);
}
